// Head_44178033607381
// MI455X (gfx1250) — matrix operand order repaired and checked by reading, not compiled, not run on hardware
//
#include <hip/hip_runtime.h>

typedef __attribute__((ext_vector_type(16))) _Float16 v16h;
typedef __attribute__((ext_vector_type(8)))  float    v8f;
typedef unsigned int u32x4 __attribute__((ext_vector_type(4)));
typedef int          i32x8 __attribute__((ext_vector_type(8)));
typedef int          i32x4 __attribute__((ext_vector_type(4)));

static constexpr int   Tt    = 128;
static constexpr int   Ee    = 192;
static constexpr int   Hh    = 192;
static constexpr float SCALE = 0.0721687836487032f; // 192^-0.5 (reference scales by E^-0.5)

// LDS strides (halves); all row starts 16B-aligned for b128 LDS traffic
static constexpr int XSld = 200; // x(f16) -> Q scratch -> P   [T][*]
static constexpr int KRld = 200; // K row-major [T][H]
static constexpr int VTld = 136; // V^T        [H][T]
static constexpr int WSld = 200; // staged weight (f16) [H][E]

static constexpr int OFF_XH = 0;
static constexpr int OFF_KR = OFF_XH + Tt * XSld * 2;  // 51200
static constexpr int OFF_VT = OFF_KR + Tt * KRld * 2;  // 102400
static constexpr int OFF_WB = OFF_VT + Hh * VTld * 2;  // 154624
static constexpr int SMEM_BYTES = OFF_WB + Hh * WSld * 2; // 231424 (~226KB)

__device__ __forceinline__ v8f wmma_f16(const v16h& a, const v16h& b, const v8f& c) {
  return __builtin_amdgcn_wmma_f32_16x16x32_f16(false, a, false, b, (short)0, c, false, false);
}

// A fragment (16x32): lane lo = row; two contiguous 16B runs -> 2x ds_load_b128
__device__ __forceinline__ v16h load_A(const _Float16* buf, int ld, int row0, int k0,
                                       int lo, int hi) {
  union { u32x4 q[2]; v16h v; } f;
  const u32x4* p = (const u32x4*)(buf + (size_t)(row0 + lo) * ld + k0 + 16 * hi);
  f.q[0] = p[0]; // halves 0..7   (K = 16*hi + 0..7)
  f.q[1] = p[1]; // halves 8..15  (K = 16*hi + 8..15)
  return f.v;
}

// B fragment (32x16) where Bmat[k][n] has k contiguous in the LDS row of index n:
// b[idx] = buf[(n0+lo)*ld + k0 + 16*hi + idx]  -> 2x ds_load_b128
__device__ __forceinline__ v16h load_Bc(const _Float16* buf, int ld, int n0, int k0,
                                        int lo, int hi) {
  union { u32x4 q[2]; v16h v; } f;
  const u32x4* p = (const u32x4*)(buf + (size_t)(n0 + lo) * ld + k0 + 16 * hi);
  f.q[0] = p[0];
  f.q[1] = p[1];
  return f.v;
}

// stage one f32 weight [192][192] -> f16 LDS [192][WSld] (block-wide, coalesced)
__device__ __forceinline__ void stage_w(const float* __restrict__ W, _Float16* wb, int tid) {
  const float4* w4 = (const float4*)W;
#pragma unroll
  for (int i = 0; i < (Hh * Ee / 4) / 256; ++i) { // 36
    int    idx = tid + 256 * i;
    float4 f   = w4[idx];
    int    row = idx / (Ee / 4);
    int    col = (idx % (Ee / 4)) * 4;
    _Float16* d = &wb[row * WSld + col];
    d[0] = (_Float16)f.x; d[1] = (_Float16)f.y;
    d[2] = (_Float16)f.z; d[3] = (_Float16)f.w;
  }
}

__global__ __launch_bounds__(256, 1) void attn_head_kernel(
    const float* __restrict__ x, const float* __restrict__ Wq,
    const float* __restrict__ Wk, const float* __restrict__ Wv,
    float* __restrict__ out, int B) {
  __shared__ __align__(16) unsigned char smem[SMEM_BYTES];
  _Float16* xh = (_Float16*)(smem + OFF_XH);
  _Float16* kr = (_Float16*)(smem + OFF_KR);
  _Float16* vt = (_Float16*)(smem + OFF_VT);
  _Float16* wb = (_Float16*)(smem + OFF_WB);
  float*    xr = (float*)(smem + OFF_KR); // f32 x staging overlays Kr+V^T (dead until later)

  const int b    = blockIdx.x;
  const int tid  = threadIdx.x;
  const int lane = tid & 31;
  const int wv   = tid >> 5; // 0..7 : owned query row-tile
  const int lo   = lane & 15;
  const int hi   = lane >> 4;
  const int mt   = wv;

  const float* xg = x + (size_t)b * Tt * Ee;

  // ---- TDM: DMA x[b] (128x192 f32, contiguous 2D tile) into LDS overlay ----
  if (wv == 0) {
    unsigned long long ga = (unsigned long long)(const void*)xg;
    unsigned ldsOff = (unsigned)(unsigned long long)(const void*)xr; // flat[31:0] = LDS offset
    u32x4 g0 = {
        1u,                                                // count=1, user descriptor
        ldsOff,                                            // lds_addr
        (unsigned)ga,                                      // global_addr[31:0]
        (unsigned)((ga >> 32) & 0x01FFFFFFu) | 0x80000000u // global_addr[56:32] | type=2
    };
    i32x8 g1 = {
        (int)(2u << 16),                  // workgroup_mask=0 | data_size=2 (4B)
        (int)((unsigned)Ee << 16),        // tensor_dim0[15:0]=192 in [31:16]
        (int)((unsigned)Tt << 16),        // tensor_dim0 hi=0 | tensor_dim1[15:0]=128
        (int)((unsigned)Ee << 16),        // tensor_dim1 hi=0 | tile_dim0=192
        (int)Tt,                          // tile_dim1=128 | tile_dim2=0
        (int)Ee,                          // tensor_dim0_stride[31:0]=192
        (int)((unsigned)(Tt * Ee) << 16), // dim0_stride hi=0 | tensor_dim1_stride lo16
        0                                 // tensor_dim1_stride hi
    };
    i32x4 z4 = {0, 0, 0, 0};
    i32x8 z8 = {0, 0, 0, 0, 0, 0, 0, 0};
    __builtin_amdgcn_tensor_load_to_lds(g0, g1, z4, z4, z8, 0);
  }
  // while TDM streams x, all threads stage Wq (global f32 -> f16 LDS)
  stage_w(Wq, wb, tid);
  if (b + 1 < B) __builtin_prefetch(xg + (size_t)Tt * Ee + tid * 64, 0, 0);
  if (wv == 0) __builtin_amdgcn_s_wait_tensorcnt(0);
  __syncthreads(); // x f32 in xr, Wq staged

  // ---- convert x f32 -> f16 xh (block-wide, ds b128 in / b64 out) ----
  {
    const float4* xr4 = (const float4*)xr;
#pragma unroll
    for (int i = 0; i < (Tt * Ee / 4) / 256; ++i) { // 24
      int    idx = tid + 256 * i;
      float4 f   = xr4[idx];
      int    row = idx / (Ee / 4);
      int    col = (idx % (Ee / 4)) * 4;
      _Float16* d = &xh[row * XSld + col];
      d[0] = (_Float16)f.x; d[1] = (_Float16)f.y;
      d[2] = (_Float16)f.z; d[3] = (_Float16)f.w;
    }
  }
  __syncthreads(); // xh ready; xr region now dead -> Kr/V^T may be written

  // x A-fragments for this wave's 16 rows (shared by all three projections)
  v16h av[6];
#pragma unroll
  for (int ks = 0; ks < 6; ++ks) av[ks] = load_A(xh, XSld, 16 * mt, 32 * ks, lo, hi);

  // ---- Q = x Wq^T (keep in registers) ----
  v8f qacc[12];
#pragma unroll
  for (int nt = 0; nt < 12; ++nt) {
    v8f acc = {};
#pragma unroll
    for (int ks = 0; ks < 6; ++ks)
      acc = wmma_f16(av[ks], load_Bc(wb, WSld, 16 * nt, 32 * ks, lo, hi), acc);
    qacc[nt] = acc;
  }
  __syncthreads(); // done reading Wq

  // ---- K = x Wk^T -> store row-major (B-fragment-contiguous for S) ----
  stage_w(Wk, wb, tid);
  __syncthreads();
#pragma unroll
  for (int nt = 0; nt < 12; ++nt) {
    v8f acc = {};
#pragma unroll
    for (int ks = 0; ks < 6; ++ks)
      acc = wmma_f16(av[ks], load_Bc(wb, WSld, 16 * nt, 32 * ks, lo, hi), acc);
#pragma unroll
    for (int r = 0; r < 8; ++r)
      kr[(16 * mt + 8 * hi + r) * KRld + 16 * nt + lo] = (_Float16)acc[r];
  }
  __syncthreads(); // done reading Wk

  // ---- V = x Wv^T -> store transposed (B-fragment-contiguous for O) ----
  stage_w(Wv, wb, tid);
  __syncthreads();
#pragma unroll
  for (int nt = 0; nt < 12; ++nt) {
    v8f acc = {};
#pragma unroll
    for (int ks = 0; ks < 6; ++ks)
      acc = wmma_f16(av[ks], load_Bc(wb, WSld, 16 * nt, 32 * ks, lo, hi), acc);
#pragma unroll
    for (int r = 0; r < 8; ++r)
      vt[(16 * nt + lo) * VTld + 16 * mt + 8 * hi + r] = (_Float16)acc[r];
  }

  // ---- rewrite own x rows with Q (f16), re-read as A-fragments ----
#pragma unroll
  for (int nt = 0; nt < 12; ++nt)
#pragma unroll
    for (int r = 0; r < 8; ++r)
      xh[(16 * mt + 8 * hi + r) * XSld + 16 * nt + lo] = (_Float16)qacc[nt][r];

  asm volatile("s_wait_dscnt 0x0" ::: "memory"); // in-wave LDS RAW fence

  v16h qa[6];
#pragma unroll
  for (int ks = 0; ks < 6; ++ks) qa[ks] = load_A(xh, XSld, 16 * mt, 32 * ks, lo, hi);

  // zero P region (cols 0..127 of own rows; Q fragments already in registers)
  for (int i = lane; i < 16 * 64; i += 32) {
    int rr = i >> 6, cc = i & 63;
    *(unsigned int*)&xh[(16 * mt + rr) * XSld + 2 * cc] = 0u;
  }

  __syncthreads(); // Kr / V^T visible to all waves

  // ---- S = Q K^T over causal tiles (wave-uniform guards keep EXEC full) ----
  v8f s[8];
#pragma unroll
  for (int ntk = 0; ntk < 8; ++ntk) {
    if (ntk <= mt) {
      v8f acc = {};
#pragma unroll
      for (int ks = 0; ks < 6; ++ks)
        acc = wmma_f16(qa[ks], load_Bc(kr, KRld, 16 * ntk, 32 * ks, lo, hi), acc);
      s[ntk] = acc;
    }
  }

  // ---- causal mask + softmax in registers (16-lane shfl reductions per row) ----
#pragma unroll
  for (int r = 0; r < 8; ++r) {
    const int row = 16 * mt + 8 * hi + r;
    float     mx  = -3.0e38f;
#pragma unroll
    for (int ntk = 0; ntk < 8; ++ntk)
      if (ntk <= mt) {
        bool keep = (16 * ntk + lo) <= row;
        mx = keep ? fmaxf(mx, s[ntk][r]) : mx;
      }
#pragma unroll
    for (int off = 1; off < 16; off <<= 1)
      mx = fmaxf(mx, __shfl_xor(mx, off, 32));
    float sum = 0.f;
#pragma unroll
    for (int ntk = 0; ntk < 8; ++ntk)
      if (ntk <= mt) {
        bool  keep = (16 * ntk + lo) <= row;
        float pv   = keep ? __expf(SCALE * (s[ntk][r] - mx)) : 0.f;
        s[ntk][r]  = pv;
        sum += pv;
      }
#pragma unroll
    for (int off = 1; off < 16; off <<= 1)
      sum += __shfl_xor(sum, off, 32);
    float inv = 1.0f / sum;
#pragma unroll
    for (int ntk = 0; ntk < 8; ++ntk)
      if (ntk <= mt)
        xh[row * XSld + 16 * ntk + lo] = (_Float16)(s[ntk][r] * inv);
  }

  asm volatile("s_wait_dscnt 0x0" ::: "memory");

  // ---- O = P V (skip all-zero K-steps beyond the causal frontier) ----
  v8f o[12];
#pragma unroll
  for (int nt = 0; nt < 12; ++nt) { v8f z = {}; o[nt] = z; }
#pragma unroll
  for (int ks = 0; ks < 4; ++ks) {
    if (ks <= (mt >> 1)) {
      v16h pa = load_A(xh, XSld, 16 * mt, 32 * ks, lo, hi);
#pragma unroll
      for (int nt = 0; nt < 12; ++nt)
        o[nt] = wmma_f16(pa, load_Bc(vt, VTld, 16 * nt, 32 * ks, lo, hi), o[nt]);
    }
  }

  float* og = out + (size_t)b * Tt * Hh;
#pragma unroll
  for (int nt = 0; nt < 12; ++nt)
#pragma unroll
    for (int r = 0; r < 8; ++r)
      og[(size_t)(16 * mt + 8 * hi + r) * Hh + 16 * nt + lo] = o[nt][r];
}

extern "C" void kernel_launch(void* const* d_in, const int* in_sizes, int n_in,
                              void* d_out, int out_size, void* d_ws, size_t ws_size,
                              hipStream_t stream) {
  const float* x   = (const float*)d_in[0];
  const float* Wq  = (const float*)d_in[1];
  const float* Wk  = (const float*)d_in[2];
  const float* Wv  = (const float*)d_in[3];
  float*       out = (float*)d_out;
  const int    B   = in_sizes[0] / (Tt * Ee);
  attn_head_kernel<<<dim3(B), dim3(256), 0, stream>>>(x, Wq, Wk, Wv, out, B);
}
